// LongTermMemoryModule_82489141887465
// MI455X (gfx1250) — matrix operand order repaired and checked by reading, not compiled, not run on hardware
//
#include <hip/hip_runtime.h>
#include <hip/hip_bf16.h>

// ---------------------------------------------------------------------------
// Types for CDNA5 WMMA (wave32). D = A(16x32 bf16) x B(32x16 bf16) + C(f32).
// ---------------------------------------------------------------------------
typedef __attribute__((ext_vector_type(16))) __bf16 v16bf;
typedef __attribute__((ext_vector_type(8)))  float  v8f;

static __device__ __forceinline__ v8f wmma_bf16(v16bf a, v16bf b, v8f c) {
    // (neg_a, A, neg_b, B, c_mod, C, reuse_a, reuse_b)
    return __builtin_amdgcn_wmma_f32_16x16x32_bf16(false, a, false, b,
                                                   (short)0, c, false, false);
}

// A-matrix fragment (16 rows x 32 K, bf16) from LDS, row-major with `stride`
// elements per row. Per ISA layout: lane<16 holds row=lane, K {0..7,16..23};
// lane>=16 holds row=lane-16, K {8..15,24..31}.
static __device__ __forceinline__ v16bf load_frag_a(const __bf16* base, int r0,
                                                    int stride, int k0, int lane) {
    int row = r0 + (lane & 15);
    int kb  = k0 + ((lane >> 4) << 3);              // 0 or 8
    const __bf16* p = base + row * stride + kb;
    union { v16bf v; uint4 q[2]; } u;
    u.q[0] = *(const uint4*)(p);                    // K kb .. kb+7
    u.q[1] = *(const uint4*)(p + 16);               // K kb+16 .. kb+23
    return u.v;
}

// B-matrix fragment (32 K x 16 cols, bf16). LDS holds B transposed: row = col n,
// contiguous K. Per ISA layout: lane<16 holds col=lane K {0..7,16..23}; lane>=16 K {8..15,24..31}.
static __device__ __forceinline__ v16bf load_frag_b(const __bf16* base, int c0,
                                                    int stride, int k0, int lane) {
    int col = c0 + (lane & 15);
    int kb  = k0 + ((lane >> 4) << 3);              // 0 or 8
    const __bf16* p = base + col * stride + kb;
    union { v16bf v; uint4 q[2]; } u;
    u.q[0] = *(const uint4*)(p);                    // K kb .. kb+7
    u.q[1] = *(const uint4*)(p + 16);               // K kb+16 .. kb+23
    return u.v;
}

// ---------------------------------------------------------------------------
// GEMM: C[M x 512] = alpha * (A[M x 512] @ W[512 x 512]^T + bias)
// Block tile 128x64, 8 waves of 32x32 (2x2 WMMA tiles). K staged in LDS as bf16.
// ---------------------------------------------------------------------------
__global__ __launch_bounds__(256)
void ltm_gemm_bf16_wmma(const float* __restrict__ A, const float* __restrict__ W,
                        const float* __restrict__ bias, float* __restrict__ C,
                        float alpha) {
    constexpr int K = 512;
    __shared__ __attribute__((aligned(16))) __bf16 As[128 * 32];
    __shared__ __attribute__((aligned(16))) __bf16 Bs[64 * 32];

    const int t    = threadIdx.x;
    const int lane = t & 31;
    const int w    = t >> 5;
    const int wm   = w & 3;        // 0..3 -> 32-row strip
    const int wn   = w >> 2;       // 0..1 -> 32-col strip
    const int mBase = blockIdx.x * 128;
    const int nBase = blockIdx.y * 64;

    v8f c00 = {}, c01 = {}, c10 = {}, c11 = {};

    const int arow = t >> 1, aseg = (t & 1) * 16;   // 2 thr/row, 16 f32 each
    const int brow = t >> 2, bseg = (t & 3) * 8;    // 4 thr/row, 8 f32 each

    for (int k0 = 0; k0 < K; k0 += 32) {
        const float* ap = A + (size_t)(mBase + arow) * K + k0 + aseg;
#pragma unroll
        for (int i = 0; i < 4; ++i) {
            float4 f = *(const float4*)(ap + 4 * i);
            int d = arow * 32 + aseg + 4 * i;
            As[d + 0] = (__bf16)f.x; As[d + 1] = (__bf16)f.y;
            As[d + 2] = (__bf16)f.z; As[d + 3] = (__bf16)f.w;
        }
        const float* bp = W + (size_t)(nBase + brow) * K + k0 + bseg;
#pragma unroll
        for (int i = 0; i < 2; ++i) {
            float4 f = *(const float4*)(bp + 4 * i);
            int d = brow * 32 + bseg + 4 * i;
            Bs[d + 0] = (__bf16)f.x; Bs[d + 1] = (__bf16)f.y;
            Bs[d + 2] = (__bf16)f.z; Bs[d + 3] = (__bf16)f.w;
        }
        __syncthreads();

        v16bf a0 = load_frag_a(As, wm * 32,      32, 0, lane);
        v16bf a1 = load_frag_a(As, wm * 32 + 16, 32, 0, lane);
        v16bf b0 = load_frag_b(Bs, wn * 32,      32, 0, lane);
        v16bf b1 = load_frag_b(Bs, wn * 32 + 16, 32, 0, lane);
        c00 = wmma_bf16(a0, b0, c00);
        c01 = wmma_bf16(a0, b1, c01);
        c10 = wmma_bf16(a1, b0, c10);
        c11 = wmma_bf16(a1, b1, c11);
        __syncthreads();
    }

    // Epilogue: D layout -> VGPR p: lanes0-15 M=p, lanes16-31 M=8+p; N=lane&15.
    const int rowoff = (lane & 16) ? 8 : 0;
    const int col0   = nBase + wn * 32 + (lane & 15);
    const float b0v = bias[col0], b1v = bias[col0 + 16];
#pragma unroll
    for (int p = 0; p < 8; ++p) {
        int r0 = mBase + wm * 32 + rowoff + p;
        int r1 = r0 + 16;
        C[(size_t)r0 * 512 + col0]      = alpha * (c00[p] + b0v);
        C[(size_t)r0 * 512 + col0 + 16] = alpha * (c01[p] + b1v);
        C[(size_t)r1 * 512 + col0]      = alpha * (c10[p] + b0v);
        C[(size_t)r1 * 512 + col0 + 16] = alpha * (c11[p] + b1v);
    }
}

// ---------------------------------------------------------------------------
// Flash attention per (b,h): Q(64x64, pre-scaled) vs K/V (8192x64), online
// softmax, O = softmax(QK^T) @ V. One 256-thread block per (b,h).
// ---------------------------------------------------------------------------
__global__ __launch_bounds__(256)
void ltm_flash_attn_wmma(const float* __restrict__ qw, const float* __restrict__ kw,
                         const float* __restrict__ vw, float* __restrict__ ow) {
    constexpr int D = 512, S = 64, HD = 64, MT = 64, MTOT = 8192;
    __shared__ __attribute__((aligned(16))) __bf16 Qs[S * HD];
    __shared__ __attribute__((aligned(16))) __bf16 Ks[MT * HD];
    __shared__ __attribute__((aligned(16))) __bf16 Vs[HD * MT];   // transposed
    __shared__ __attribute__((aligned(16))) __bf16 Ps[S * MT];
    __shared__ __attribute__((aligned(16))) float  Ss[S * MT];
    __shared__ float m_lds[S], l_lds[S], corr_lds[S];

    const int b = blockIdx.x >> 3;
    const int h = blockIdx.x & 7;
    const int t    = threadIdx.x;
    const int lane = t & 31;
    const int w    = t >> 5;
    const int wm   = w & 3;        // 16-row S strip
    const int wn   = w >> 2;       // 32-col strip (m for scores, d for output)
    const int r0   = wm * 16;
    const int c0   = wn * 32;
    const int rowoff = (lane & 16) ? 8 : 0;

    // Load Q tile (already scaled by 1/sqrt(hd) in the projection GEMM).
    {
        int row = t >> 2, seg = (t & 3) * 16;
        const float* qp = qw + (size_t)(b * S + row) * D + h * HD + seg;
#pragma unroll
        for (int i = 0; i < 4; ++i) {
            float4 f = *(const float4*)(qp + 4 * i);
            int d = row * HD + seg + 4 * i;
            Qs[d + 0] = (__bf16)f.x; Qs[d + 1] = (__bf16)f.y;
            Qs[d + 2] = (__bf16)f.z; Qs[d + 3] = (__bf16)f.w;
        }
    }
    if (t < S) { m_lds[t] = -1e30f; l_lds[t] = 0.0f; }
    __syncthreads();

    v8f o0 = {}, o1 = {};

    for (int m0 = 0; m0 < MTOT; m0 += MT) {
        // Stage K tile (row-major) and V tile (transposed) as bf16.
        {
            int row = t >> 2, seg = (t & 3) * 16;
            const float* kp = kw + (size_t)(m0 + row) * D + h * HD + seg;
            const float* vp = vw + (size_t)(m0 + row) * D + h * HD + seg;
#pragma unroll
            for (int i = 0; i < 4; ++i) {
                float4 f = *(const float4*)(kp + 4 * i);
                int d = row * HD + seg + 4 * i;
                Ks[d + 0] = (__bf16)f.x; Ks[d + 1] = (__bf16)f.y;
                Ks[d + 2] = (__bf16)f.z; Ks[d + 3] = (__bf16)f.w;
                float4 g = *(const float4*)(vp + 4 * i);
                int dc = seg + 4 * i;
                Vs[(dc + 0) * MT + row] = (__bf16)g.x;
                Vs[(dc + 1) * MT + row] = (__bf16)g.y;
                Vs[(dc + 2) * MT + row] = (__bf16)g.z;
                Vs[(dc + 3) * MT + row] = (__bf16)g.w;
            }
        }
        __syncthreads();

        // S tile = Q @ K^T : 64x64, each wave 16x32, K-dim = hd = 64.
        v8f s0 = {}, s1 = {};
#pragma unroll
        for (int k0 = 0; k0 < HD; k0 += 32) {
            v16bf aq = load_frag_a(Qs, r0, HD, k0, lane);
            v16bf b0 = load_frag_b(Ks, c0,      HD, k0, lane);
            v16bf b1 = load_frag_b(Ks, c0 + 16, HD, k0, lane);
            s0 = wmma_bf16(aq, b0, s0);
            s1 = wmma_bf16(aq, b1, s1);
        }
#pragma unroll
        for (int p = 0; p < 8; ++p) {
            int row = r0 + rowoff + p;
            Ss[row * MT + c0 + (lane & 15)]      = s0[p];
            Ss[row * MT + c0 + 16 + (lane & 15)] = s1[p];
        }
        __syncthreads();

        // Per-row running max + correction factor.
        if (t < S) {
            float mx = -1e30f;
            for (int j = 0; j < MT; ++j) mx = fmaxf(mx, Ss[t * MT + j]);
            float m_old = m_lds[t];
            float m_new = fmaxf(m_old, mx);
            corr_lds[t] = __expf(m_old - m_new);
            m_lds[t]    = m_new;
        }
        __syncthreads();

        // P = exp(S - m_new) as bf16.
        {
            float mrow = m_lds[t >> 2];
            int base = t * 16;
#pragma unroll
            for (int e = 0; e < 16; ++e)
                Ps[base + e] = (__bf16)__expf(Ss[base + e] - mrow);
        }
        __syncthreads();

        // l update (reads only Ps/corr; no hazard with the WMMA phase below).
        if (t < S) {
            float s = 0.0f;
            for (int j = 0; j < MT; ++j) s += (float)Ps[t * MT + j];
            l_lds[t] = l_lds[t] * corr_lds[t] + s;
        }

        // Rescale O accumulators, then O += P @ V (K-dim = m tile = 64).
#pragma unroll
        for (int p = 0; p < 8; ++p) {
            float f = corr_lds[r0 + rowoff + p];
            o0[p] *= f;
            o1[p] *= f;
        }
#pragma unroll
        for (int k0 = 0; k0 < MT; k0 += 32) {
            v16bf apv = load_frag_a(Ps, r0, MT, k0, lane);
            v16bf bv0 = load_frag_b(Vs, c0,      MT, k0, lane);
            v16bf bv1 = load_frag_b(Vs, c0 + 16, MT, k0, lane);
            o0 = wmma_bf16(apv, bv0, o0);
            o1 = wmma_bf16(apv, bv1, o1);
        }
        __syncthreads();
    }

    // Normalize and write (B,S,H,hd) flattened as (B*S, 512).
    const int col = h * HD + c0 + (lane & 15);
#pragma unroll
    for (int p = 0; p < 8; ++p) {
        int row = r0 + rowoff + p;
        float inv = 1.0f / l_lds[row];
        ow[(size_t)(b * S + row) * D + col]      = o0[p] * inv;
        ow[(size_t)(b * S + row) * D + col + 16] = o1[p] * inv;
    }
}

// ---------------------------------------------------------------------------
// new_bank = memory_bank with rows (ptr+i)%M, i<2048 replaced by memory rows.
// ptr read on-device (no host sync allowed under graph capture).
// ---------------------------------------------------------------------------
__global__ __launch_bounds__(256)
void ltm_store_bank(const float* __restrict__ memory, const float* __restrict__ bank,
                    const int* __restrict__ ptr_p, float* __restrict__ out) {
    constexpr int M = 8192, D = 512, N = 2048;
    int i = (blockIdx.x * blockDim.x + threadIdx.x) * 4;   // float4 granularity
    int r = i >> 9, c = i & (D - 1);
    int p = *ptr_p;
    int src = r - p; if (src < 0) src += M;
    float4 v;
    if (src < N) v = *(const float4*)(memory + (size_t)src * D + c);
    else         v = *(const float4*)(bank   + (size_t)r   * D + c);
    *(float4*)(out + (size_t)r * D + c) = v;
}

// ---------------------------------------------------------------------------
extern "C" void kernel_launch(void* const* d_in, const int* in_sizes, int n_in,
                              void* d_out, int out_size, void* d_ws, size_t ws_size,
                              hipStream_t stream) {
    const float* query      = (const float*)d_in[0];   // 32*64*512
    const float* memory     = (const float*)d_in[1];   // 32*64*512
    const float* bank       = (const float*)d_in[2];   // 8192*512
    const float* in_proj_w  = (const float*)d_in[3];   // 1536*512
    const float* in_proj_b  = (const float*)d_in[4];   // 1536
    const float* out_proj_w = (const float*)d_in[5];   // 512*512
    const float* out_proj_b = (const float*)d_in[6];   // 512
    const int*   ptr_p      = (const int*)d_in[7];

    float* retrieved = (float*)d_out;                  // 2048*512
    float* new_bank  = (float*)d_out + 2048 * 512;     // 8192*512

    float* q_ws   = (float*)d_ws;                      // 2048*512
    float* k_ws   = q_ws + 2048 * 512;                 // 8192*512
    float* v_ws   = k_ws + 8192 * 512;                 // 8192*512
    float* att_ws = v_ws + 8192 * 512;                 // 2048*512

    const float scale = 0.125f;                        // 1/sqrt(64)

    // Projections (scale folded into Q).
    ltm_gemm_bf16_wmma<<<dim3(16, 8), 256, 0, stream>>>(
        query, in_proj_w, in_proj_b, q_ws, scale);
    ltm_gemm_bf16_wmma<<<dim3(64, 8), 256, 0, stream>>>(
        bank, in_proj_w + 512 * 512, in_proj_b + 512, k_ws, 1.0f);
    ltm_gemm_bf16_wmma<<<dim3(64, 8), 256, 0, stream>>>(
        bank, in_proj_w + 2 * 512 * 512, in_proj_b + 1024, v_ws, 1.0f);

    // Flash attention: one block per (b,h).
    ltm_flash_attn_wmma<<<dim3(32 * 8), 256, 0, stream>>>(q_ws, k_ws, v_ws, att_ws);

    // Output projection.
    ltm_gemm_bf16_wmma<<<dim3(16, 8), 256, 0, stream>>>(
        att_ws, out_proj_w, out_proj_b, retrieved, 1.0f);

    // Bank splice.
    ltm_store_bank<<<dim3(8192 * 512 / 4 / 256), 256, 0, stream>>>(
        memory, bank, ptr_p, new_bank);
}
